// Attention_5076651344278
// MI455X (gfx1250) — matrix operand order repaired and checked by reading, not compiled, not run on hardware
//
#include <hip/hip_runtime.h>

#define B_    2
#define L_    2048
#define E_    1024
#define H_    16
#define DKV_  64
#define NOUT_ 3072   // 3*H*DKV, row stride of W_qkv
#define NQK_  2048   // H * 2 * DKV  (Q and K columns only)

typedef __attribute__((ext_vector_type(16))) __bf16 v16bf;
typedef __attribute__((ext_vector_type(8)))  float  v8f;

union Frag { unsigned u[8]; v16bf v; };

static __device__ __forceinline__ unsigned short f2bf(float f) {
  unsigned u = __builtin_bit_cast(unsigned, f);
  u += 0x7FFFu + ((u >> 16) & 1u);        // round-to-nearest-even
  return (unsigned short)(u >> 16);
}
static __device__ __forceinline__ unsigned pack2(float lo, float hi) {
  return (unsigned)f2bf(lo) | ((unsigned)f2bf(hi) << 16);
}
static __device__ __forceinline__ int colmap(int n) {
  // n in [0,2048): h = n>>7, qk = (n>>6)&1, d = n&63 -> W_qkv column
  return (n >> 7) * 192 + (((n >> 6) & 1) << 6) + (n & 63);
}

// -----------------------------------------------------------------------------
// Pre-pass 1: X (f32) -> Xb (bf16), 8 elements per thread.
// -----------------------------------------------------------------------------
__global__ __launch_bounds__(256) void cvt_x_kernel(
    const float* __restrict__ X, unsigned short* __restrict__ Xb) {
  const size_t t = (size_t)blockIdx.x * 256 + threadIdx.x;   // 524288 threads
  const float4* src = (const float4*)X;
  uint2* dst = (uint2*)Xb;
  float4 v0 = src[t * 2];
  float4 v1 = src[t * 2 + 1];
  dst[t * 2]     = make_uint2(pack2(v0.x, v0.y), pack2(v0.z, v0.w));
  dst[t * 2 + 1] = make_uint2(pack2(v1.x, v1.y), pack2(v1.z, v1.w));
}

// -----------------------------------------------------------------------------
// Pre-pass 2: gather Q/K columns of W_qkv, transpose into WbT[n][k] (bf16).
// -----------------------------------------------------------------------------
__global__ __launch_bounds__(256) void cvt_w_kernel(
    const float* __restrict__ W, unsigned short* __restrict__ WbT) {
  const int t = blockIdx.x * 256 + threadIdx.x;    // 2048 * 128 threads
  const int n  = t >> 7;
  const int k0 = (t & 127) << 3;
  const int col = colmap(n);
  uint4 o;
  o.x = pack2(W[(size_t)(k0 + 0) * NOUT_ + col], W[(size_t)(k0 + 1) * NOUT_ + col]);
  o.y = pack2(W[(size_t)(k0 + 2) * NOUT_ + col], W[(size_t)(k0 + 3) * NOUT_ + col]);
  o.z = pack2(W[(size_t)(k0 + 4) * NOUT_ + col], W[(size_t)(k0 + 5) * NOUT_ + col]);
  o.w = pack2(W[(size_t)(k0 + 6) * NOUT_ + col], W[(size_t)(k0 + 7) * NOUT_ + col]);
  *(uint4*)(WbT + (size_t)n * E_ + k0) = o;
}

// -----------------------------------------------------------------------------
// Kernel A: Y = Xb @ WbT^T (+bias), bf16 WMMA. One wave computes a 16x64
// strip; k-steps are software-pipelined (double-buffered fragments) so the
// next step's global_load_b128 clauses are in flight during this step's WMMAs.
// -----------------------------------------------------------------------------
struct KStep { Frag a; Frag b[4]; };

__global__ __launch_bounds__(128) void qk_proj_kernel(
    const unsigned short* __restrict__ Xb,
    const unsigned short* __restrict__ WbT,
    const float* __restrict__ bias,
    unsigned short* __restrict__ Qbuf, unsigned short* __restrict__ Kbuf) {
  const int lane = threadIdx.x & 31;
  const int wave = threadIdx.x >> 5;
  const int task = blockIdx.x * 4 + wave;     // 256 M-tiles * 32 N-groups
  const int mt = task >> 5;                   // M tile (16 rows)
  const int ng = task & 31;                   // N group (64 cols)
  const int m0 = mt << 4, n0 = ng << 6;

  const int r    = lane & 15;                 // A row / B,C column within tile
  const int half = lane >> 4;
  const int kb   = half << 4;                 // unused; B-frag K offsets use 8*half (table order)

  const unsigned* arow = (const unsigned*)(Xb + (size_t)(m0 + r) * E_);
  const unsigned* brow0 = (const unsigned*)(WbT + (size_t)(n0 + r) * E_);
  const unsigned* brow1 = (const unsigned*)(WbT + (size_t)(n0 + 16 + r) * E_);
  const unsigned* brow2 = (const unsigned*)(WbT + (size_t)(n0 + 32 + r) * E_);
  const unsigned* brow3 = (const unsigned*)(WbT + (size_t)(n0 + 48 + r) * E_);

  auto load_step = [&](KStep& s, int k0) {
#pragma unroll
    for (int v = 0; v < 8; ++v) {
      // 16-bit A 16x32 layout: VGPR v holds K = base(v) + 8*half + {0,1}
      const int kk = ((v & 4) << 2) + ((v & 3) << 1) + (half << 3);
      s.a.u[v] = arow[(k0 + kk) >> 1];
    }
#pragma unroll
    for (int v = 0; v < 8; ++v) {
      const int kv = (k0 + ((v & 4) << 2) + ((v & 3) << 1) + (half << 3)) >> 1;
      s.b[0].u[v] = brow0[kv];
      s.b[1].u[v] = brow1[kv];
      s.b[2].u[v] = brow2[kv];
      s.b[3].u[v] = brow3[kv];
    }
  };

  KStep s0, s1;
  v8f c0 = {}, c1 = {}, c2 = {}, c3 = {};
  load_step(s0, 0);
  for (int k0 = 0; k0 < E_; k0 += 64) {
    load_step(s1, k0 + 32);                        // in flight during WMMAs
    c0 = __builtin_amdgcn_wmma_f32_16x16x32_bf16(false, s0.a.v, false, s0.b[0].v,
                                                 (short)0, c0, false, false);
    c1 = __builtin_amdgcn_wmma_f32_16x16x32_bf16(false, s0.a.v, false, s0.b[1].v,
                                                 (short)0, c1, false, false);
    c2 = __builtin_amdgcn_wmma_f32_16x16x32_bf16(false, s0.a.v, false, s0.b[2].v,
                                                 (short)0, c2, false, false);
    c3 = __builtin_amdgcn_wmma_f32_16x16x32_bf16(false, s0.a.v, false, s0.b[3].v,
                                                 (short)0, c3, false, false);
    if (k0 + 64 < E_) load_step(s0, k0 + 64);      // in flight during WMMAs
    c0 = __builtin_amdgcn_wmma_f32_16x16x32_bf16(false, s1.a.v, false, s1.b[0].v,
                                                 (short)0, c0, false, false);
    c1 = __builtin_amdgcn_wmma_f32_16x16x32_bf16(false, s1.a.v, false, s1.b[1].v,
                                                 (short)0, c1, false, false);
    c2 = __builtin_amdgcn_wmma_f32_16x16x32_bf16(false, s1.a.v, false, s1.b[2].v,
                                                 (short)0, c2, false, false);
    c3 = __builtin_amdgcn_wmma_f32_16x16x32_bf16(false, s1.a.v, false, s1.b[3].v,
                                                 (short)0, c3, false, false);
  }

  v8f acc[4] = {c0, c1, c2, c3};
#pragma unroll
  for (int s = 0; s < 4; ++s) {
    const int gn  = n0 + (s << 4) + r;        // global output column
    const int qk  = (gn >> 6) & 1;
    const int h   = gn >> 7;
    const int d   = gn & 63;
    unsigned short* dst = qk ? Kbuf : Qbuf;
    const float bv = bias[colmap(gn)];
#pragma unroll
    for (int g = 0; g < 8; ++g) {
      // f32 C/D layout: VGPR g -> M = g + 8*half, N = lane&15
      const int m    = m0 + g + (half << 3);
      const int bidx = m >> 11;               // batch
      const int l    = m & (L_ - 1);          // sequence position
      dst[(((size_t)bidx * H_ + h) * L_ + l) * DKV_ + d] = f2bf(acc[s][g] + bv);
    }
  }
}

// -----------------------------------------------------------------------------
// Kernel B: per (b,h) and 16-row block: S = Q Kt / 8 (two 16x16x32 bf16 WMMA
// per key tile, double-buffered), 16x2048 f32 strip in 128 KB LDS, 256-thread
// cooperative softmax, b128-vectorized coalesced stores.
// 8 waves/WG, 2 WGs/WGP (256 KB of the 320 KB LDS) -> 4 waves/SIMD.
// -----------------------------------------------------------------------------
__global__ __launch_bounds__(256) void attn_softmax_kernel(
    const unsigned short* __restrict__ Qbuf,
    const unsigned short* __restrict__ Kbuf,
    float* __restrict__ out) {
  extern __shared__ float S[];                // 16 * 2048 floats = 128 KB
  const int lane = threadIdx.x & 31;
  const int wave = threadIdx.x >> 5;          // 0..7
  const int bh   = blockIdx.x >> 7;           // 0..31  (b*H + h)
  const int i0   = (blockIdx.x & 127) << 4;   // query row block

  const int r    = lane & 15;
  const int half = lane >> 4;
  const int kb   = half << 4;

  // Q A-fragments (rows i0..i0+15, K 0..31 and 32..63)
  Frag a0, a1;
  {
    const unsigned* q32 =
        (const unsigned*)(Qbuf + ((size_t)bh * L_ + i0 + r) * DKV_);
#pragma unroll
    for (int v = 0; v < 8; ++v) {
      const int kk = ((v & 4) << 2) + ((v & 3) << 1) + (half << 3);
      a0.u[v] = q32[kk >> 1];
      a1.u[v] = q32[(32 + kk) >> 1];
    }
  }

  const unsigned short* kbase = Kbuf + (size_t)bh * L_ * DKV_;
  Frag b[2][2];
  auto load_b = [&](int i, int jt) {
    const unsigned* k32 =
        (const unsigned*)(kbase + (size_t)((jt << 4) + r) * DKV_);
#pragma unroll
    for (int v = 0; v < 8; ++v) {
      const int k = ((v & 4) << 2) + ((v & 3) << 1) + (half << 3);
      b[i][0].u[v] = k32[k >> 1];
      b[i][1].u[v] = k32[(32 + k) >> 1];
    }
  };

  load_b(0, wave);
  int cur = 0;
  for (int jt = wave; jt < (L_ >> 4); jt += 8) {   // uniform: 16 tiles/wave
    if (jt + 8 < (L_ >> 4)) {
      load_b(cur ^ 1, jt + 8);                     // in flight during WMMAs
      __builtin_prefetch(kbase + (size_t)(((jt + 16) << 4) + r) * DKV_, 0, 3);
    }
    v8f c = {};
    c = __builtin_amdgcn_wmma_f32_16x16x32_bf16(false, a0.v, false, b[cur][0].v,
                                                (short)0, c, false, false);
    c = __builtin_amdgcn_wmma_f32_16x16x32_bf16(false, a1.v, false, b[cur][1].v,
                                                (short)0, c, false, false);
    const int j0 = jt << 4;
#pragma unroll
    for (int g = 0; g < 8; ++g)
      S[(g + (half << 3)) * L_ + j0 + r] = c[g] * 0.125f;   // 1/sqrt(64)
    cur ^= 1;
  }
  __syncthreads();

  // Softmax: 16 threads per row, 128 contiguous elements each.
  const int row = threadIdx.x >> 4;
  const int seg = threadIdx.x & 15;
  float* Srow = S + row * L_ + (seg << 7);
  float m = -3.402823466e38f;
  for (int i = 0; i < 128; ++i) m = fmaxf(m, Srow[i]);
#pragma unroll
  for (int off = 1; off < 16; off <<= 1) m = fmaxf(m, __shfl_xor(m, off, 16));
  float sum = 0.f;
  for (int i = 0; i < 128; ++i) {
    float e = __expf(Srow[i] - m);
    Srow[i] = e;
    sum += e;
  }
#pragma unroll
  for (int off = 1; off < 16; off <<= 1) sum += __shfl_xor(sum, off, 16);
  const float inv = 1.0f / sum;

  float* orow = out + ((size_t)bh * L_ + i0 + row) * L_ + (seg << 7);
  const float4* s4 = (const float4*)Srow;
  float4* o4 = (float4*)orow;
#pragma unroll 4
  for (int i = 0; i < 32; ++i) {
    float4 v = s4[i];
    v.x *= inv; v.y *= inv; v.z *= inv; v.w *= inv;
    o4[i] = v;                                  // global_store_b128
  }
}

// -----------------------------------------------------------------------------
extern "C" void kernel_launch(void* const* d_in, const int* in_sizes, int n_in,
                              void* d_out, int out_size, void* d_ws, size_t ws_size,
                              hipStream_t stream) {
  const float* X    = (const float*)d_in[0];
  const float* W    = (const float*)d_in[1];
  const float* bias = (const float*)d_in[2];
  float* out = (float*)d_out;

  unsigned short* Xb   = (unsigned short*)d_ws;                      //  8 MB
  unsigned short* WbT  = Xb  + (size_t)B_ * L_ * E_;                 //  4 MB
  unsigned short* Qbuf = WbT + (size_t)NQK_ * E_;                    //  8 MB
  unsigned short* Kbuf = Qbuf + (size_t)B_ * H_ * L_ * DKV_;         //  8 MB

  // 1) bf16 conversion pre-passes (memory-bound, L2-resident).
  cvt_x_kernel<<<dim3((B_ * L_ * E_) / 8 / 256), dim3(256), 0, stream>>>(X, Xb);
  cvt_w_kernel<<<dim3((NQK_ * (E_ / 8)) / 256), dim3(256), 0, stream>>>(W, WbT);

  // 2) QK projection: 256 M-tiles * 32 N-groups, 4 waves (strips) per block.
  qk_proj_kernel<<<dim3((256 * 32) / 4), dim3(128), 0, stream>>>(Xb, WbT, bias,
                                                                 Qbuf, Kbuf);

  // 3) Scores + softmax: one block per (b, h, 16-row block); 128 KB LDS strip.
  attn_softmax_kernel<<<dim3(B_ * H_ * (L_ / 16)), dim3(256),
                        16 * L_ * sizeof(float), stream>>>(Qbuf, Kbuf, out);
}
